// GraphConv_Decoder_22428319219807
// MI455X (gfx1250) — compile-verified, matrix operand order checked by reading, not run on hardware
//
#include <hip/hip_runtime.h>
#include <hip/hip_bf16.h>

typedef __attribute__((ext_vector_type(16))) _Float16 v16h;
typedef __attribute__((ext_vector_type(8)))  _Float16 v8h;
typedef __attribute__((ext_vector_type(8)))  float    v8f;

// ---- Fragment loaders for V_WMMA_F32_16X16X32_F16 operands (wave32) ----
// CDNA5 16-bit operand layout: lane = M (A) or N (B) in [0,15], lane>>4
// selects the K-half; element e maps to k = k0 + (e<8 ? e : e+8) with
// k0 = kc*32 + (lane>>4)*8 — two contiguous 8-element runs per fragment.

// f32 source (converted in registers; used for X/h, read once per GEMM)
__device__ __forceinline__ v16h load_frag_f32cvt(const float* __restrict__ p, int k0) {
    const float4* q0 = (const float4*)(p + k0);
    float4 f0 = q0[0];
    float4 f1 = q0[1];
    const float4* q1 = (const float4*)(p + k0 + 16);
    float4 f2 = q1[0];
    float4 f3 = q1[1];
    v16h r;
    r[0]  = (_Float16)f0.x; r[1]  = (_Float16)f0.y; r[2]  = (_Float16)f0.z; r[3]  = (_Float16)f0.w;
    r[4]  = (_Float16)f1.x; r[5]  = (_Float16)f1.y; r[6]  = (_Float16)f1.z; r[7]  = (_Float16)f1.w;
    r[8]  = (_Float16)f2.x; r[9]  = (_Float16)f2.y; r[10] = (_Float16)f2.z; r[11] = (_Float16)f2.w;
    r[12] = (_Float16)f3.x; r[13] = (_Float16)f3.y; r[14] = (_Float16)f3.z; r[15] = (_Float16)f3.w;
    return r;
}

// f16 source (pre-converted weights; two b128 loads, zero VALU conversions)
__device__ __forceinline__ v16h load_frag_h16(const _Float16* __restrict__ p, int k0) {
    v8h lo = *(const v8h*)(p + k0);
    v8h hi = *(const v8h*)(p + k0 + 16);
    v16h r;
#pragma unroll
    for (int i = 0; i < 8; ++i) { r[i] = lo[i]; r[i + 8] = hi[i]; }
    return r;
}

// One-shot f32 -> f16 conversion of a weight matrix (tiny: <= 16K elements)
__global__ __launch_bounds__(256)
void cvt_w_f16(const float* __restrict__ src, _Float16* __restrict__ dst, int n) {
    int i = blockIdx.x * blockDim.x + threadIdx.x;
    if (i < n) dst[i] = (_Float16)src[i];
}

// Y[n][o] = sum_k X[n][k] * Wh[o][k]  (+ bias[o] if bias != nullptr)
// X: [nrows x 128] f32 row-major; Wh: [DOUT x 128] f16 row-major; Y: [nrows x DOUT].
// One wave32 owns a 16-row strip; K=128 as 4 WMMA k-steps of 32.
template <int DOUT>
__global__ __launch_bounds__(256)
void gemm_xwT(const float* __restrict__ X, const _Float16* __restrict__ Wh,
              const float* __restrict__ bias, float* __restrict__ Y, int nrows) {
    const int lane = threadIdx.x & 31;
    const int wave = threadIdx.x >> 5;
    const int tile = blockIdx.x * (blockDim.x >> 5) + wave;   // 16-row tile index
    const int row0 = tile * 16;
    if (row0 >= nrows) return;           // wave-uniform exit; EXEC stays all-1s

    const int mn = lane & 15;            // row index for A, col index for B/C/D
    const int hs = lane >> 4;            // K-half select

    // Preload all four A k-fragments for this wave's 16 rows (f32 -> f16 once).
    const float* xrow = X + (size_t)(row0 + mn) * 128;
    v16h a0 = load_frag_f32cvt(xrow, 0 * 32 + hs * 8);
    v16h a1 = load_frag_f32cvt(xrow, 1 * 32 + hs * 8);
    v16h a2 = load_frag_f32cvt(xrow, 2 * 32 + hs * 8);
    v16h a3 = load_frag_f32cvt(xrow, 3 * 32 + hs * 8);

#pragma unroll
    for (int nt = 0; nt < DOUT / 16; ++nt) {
        // B[k][n] = Wh[nt*16 + n][k]; row-major => same fragment pattern as A.
        const _Float16* wrow = Wh + (size_t)(nt * 16 + mn) * 128;
        v8f c = {};
        {
            v16h b = load_frag_h16(wrow, 0 * 32 + hs * 8);
            c = __builtin_amdgcn_wmma_f32_16x16x32_f16(false, a0, false, b, (short)0, c, false, false);
        }
        {
            v16h b = load_frag_h16(wrow, 1 * 32 + hs * 8);
            c = __builtin_amdgcn_wmma_f32_16x16x32_f16(false, a1, false, b, (short)0, c, false, false);
        }
        {
            v16h b = load_frag_h16(wrow, 2 * 32 + hs * 8);
            c = __builtin_amdgcn_wmma_f32_16x16x32_f16(false, a2, false, b, (short)0, c, false, false);
        }
        {
            v16h b = load_frag_h16(wrow, 3 * 32 + hs * 8);
            c = __builtin_amdgcn_wmma_f32_16x16x32_f16(false, a3, false, b, (short)0, c, false, false);
        }
        const float bv = bias ? bias[nt * 16 + mn] : 0.0f;
        // D layout: VGPR r, lanes 0-15 -> (M=r, N=lane), lanes 16-31 -> (M=r+8, N=lane-16)
#pragma unroll
        for (int r = 0; r < 8; ++r) {
            const int mm = r + hs * 8;
            Y[(size_t)(row0 + mm) * DOUT + nt * 16 + mn] = c[r] + bv;
        }
    }
}

// H[dst[e]][:] += T[src[e]][:] over all edges. One wave32 owns one edge:
// lane covers D/32 floats, so the row gather/scatter is one fully-coalesced
// 256/512-byte burst. Edge id is wave-uniform -> scalarize index loads and
// address math via readfirstlane.
template <int D>
__global__ __launch_bounds__(256)
void edge_scatter(const float* __restrict__ T, const int* __restrict__ ei,
                  float* __restrict__ H, int nedges) {
    constexpr int VE = D / 32;                      // floats per lane (4 or 2)
    const int lane = threadIdx.x & 31;
    const int ev = blockIdx.x * (blockDim.x >> 5) + (threadIdx.x >> 5);
    if (ev >= nedges) return;                       // wave-uniform exit
    const int e   = __builtin_amdgcn_readfirstlane(ev);
    const int src = __builtin_amdgcn_readfirstlane(ei[e]);
    const int dst = __builtin_amdgcn_readfirstlane(ei[nedges + e]);
    const float* tp = T + (size_t)src * D + lane * VE;
    float*       hp = H + (size_t)dst * D + lane * VE;
    if constexpr (VE == 4) {
        const float4 v = *(const float4*)tp;
        atomicAdd(hp + 0, v.x);
        atomicAdd(hp + 1, v.y);
        atomicAdd(hp + 2, v.z);
        atomicAdd(hp + 3, v.w);
    } else {
        const float2 v = *(const float2*)tp;
        atomicAdd(hp + 0, v.x);
        atomicAdd(hp + 1, v.y);
    }
}

extern "C" void kernel_launch(void* const* d_in, const int* in_sizes, int n_in,
                              void* d_out, int out_size, void* d_ws, size_t ws_size,
                              hipStream_t stream) {
    const float* x       = (const float*)d_in[0];
    const int*   ei      = (const int*)  d_in[1];
    const float* w_rel1  = (const float*)d_in[2];
    const float* b_rel1  = (const float*)d_in[3];
    const float* w_root1 = (const float*)d_in[4];
    const float* w_rel2  = (const float*)d_in[5];
    const float* b_rel2  = (const float*)d_in[6];
    const float* w_root2 = (const float*)d_in[7];
    float* out = (float*)d_out;

    const int N_ = in_sizes[0] / 128;   // 50000
    const int E_ = in_sizes[1] / 2;     // 600000

    // Workspace: t1 [N x 128] f32, h [N x 128] f32, t2 [N x 64] f32 (64 MB),
    // then f16 copies of the four weight matrices (96 KB).
    float* t1 = (float*)d_ws;
    float* h  = t1 + (size_t)N_ * 128;
    float* t2 = h  + (size_t)N_ * 128;
    _Float16* wh_rel1  = (_Float16*)(t2 + (size_t)N_ * 64);
    _Float16* wh_root1 = wh_rel1  + 128 * 128;
    _Float16* wh_rel2  = wh_root1 + 128 * 128;
    _Float16* wh_root2 = wh_rel2  + 64 * 128;

    // Pre-convert weights to f16 once (removes per-wave B conversions).
    cvt_w_f16<<<(128 * 128 + 255) / 256, 256, 0, stream>>>(w_rel1,  wh_rel1,  128 * 128);
    cvt_w_f16<<<(128 * 128 + 255) / 256, 256, 0, stream>>>(w_root1, wh_root1, 128 * 128);
    cvt_w_f16<<<(64  * 128 + 255) / 256, 256, 0, stream>>>(w_rel2,  wh_rel2,  64 * 128);
    cvt_w_f16<<<(64  * 128 + 255) / 256, 256, 0, stream>>>(w_root2, wh_root2, 64 * 128);

    const int tiles = (N_ + 15) / 16;           // 3125 (N is an exact multiple of 16)
    dim3 blk(256);
    dim3 grd((tiles + 7) / 8);                  // 8 waves per block, 1 tile per wave
    const int egrd = (E_ + 7) / 8;              // 8 edges (waves) per block

    // Layer 1: t1 = x @ w_rel1^T ; h = x @ w_root1^T + b_rel1
    gemm_xwT<128><<<grd, blk, 0, stream>>>(x, wh_rel1,  nullptr, t1, N_);
    gemm_xwT<128><<<grd, blk, 0, stream>>>(x, wh_root1, b_rel1,  h,  N_);
    // h[dst] += t1[src]   (lin_rel(segment_sum(x)) == segment_sum(lin_rel(x)))
    edge_scatter<128><<<egrd, blk, 0, stream>>>(t1, ei, h, E_);
    // Layer 2: t2 = h @ w_rel2^T ; out = h @ w_root2^T + b_rel2
    gemm_xwT<64><<<grd, blk, 0, stream>>>(h, wh_rel2,  nullptr, t2,  N_);
    gemm_xwT<64><<<grd, blk, 0, stream>>>(h, wh_root2, b_rel2,  out, N_);
    // out[dst] += t2[src]  (64-wide: half the edge bandwidth of aggregating h)
    edge_scatter<64><<<egrd, blk, 0, stream>>>(t2, ei, out, E_);
}
